// AdaptiveMultiScaleFusion_69148973465903
// MI455X (gfx1250) — matrix operand order repaired and checked by reading, not compiled, not run on hardware
//
#include <hip/hip_runtime.h>
#include <hip/hip_bf16.h>
#include <math.h>

typedef __attribute__((ext_vector_type(16))) _Float16 v16h;
typedef __attribute__((ext_vector_type(8)))  _Float16 v8h;
typedef __attribute__((ext_vector_type(8)))  float    v8f;

#define CIN   128
#define COUT  128
#define HH    96
#define WWID  96
#define HWP   (HH*WWID)      // 9216
#define BATCH 4
#define KK2   9
#define KDIM  (CIN*KK2)      // 1152
#define MTOT  (BATCH*HWP)    // 36864
#define BSTR  40             // padded LDS row stride (halves): 20*n mod 64 distinct for n=0..15

// ---------------- offset conv: 3x3, Cin=128 -> 18 channels ----------------
__global__ void conv_off_k(const float* __restrict__ x, const float* __restrict__ w,
                           const float* __restrict__ bias, float* __restrict__ out) {
  int gid = blockIdx.x * blockDim.x + threadIdx.x;
  if (gid >= BATCH * 18 * HWP) return;
  int p = gid % HWP; int t = gid / HWP; int oc = t % 18; int b = t / 18;
  int y = p / WWID, xx = p % WWID;
  float acc = bias[oc];
  const float* xb = x + (size_t)b * CIN * HWP;
  const float* wb = w + (size_t)oc * CIN * 9;
  for (int ki = 0; ki < 3; ++ki) {
    int yy = y + ki - 1;
    if (yy < 0 || yy >= HH) continue;
    for (int kj = 0; kj < 3; ++kj) {
      int xj = xx + kj - 1;
      if (xj < 0 || xj >= WWID) continue;
      const float* xp = xb + yy * WWID + xj;
      const float* wp = wb + ki * 3 + kj;
      #pragma unroll 4
      for (int c = 0; c < CIN; ++c)
        acc += xp[(size_t)c * HWP] * wp[(size_t)c * 9];
    }
  }
  out[gid] = acc;
}

// ------------- bilinear sampling + im2col into fp16 [M][kk*128+c] -------------
__global__ void sample_k(const float* __restrict__ x, const float* __restrict__ off,
                         _Float16* __restrict__ Ah) {
  int gid = blockIdx.x * blockDim.x + threadIdx.x;
  if (gid >= BATCH * KK2 * HWP) return;
  int p = gid % HWP; int t = gid / HWP; int kk = t % KK2; int b = t / KK2;
  int y = p / WWID, x0c = p % WWID;
  int ki = kk / 3, kj = kk % 3;
  const float* ob = off + (size_t)b * 18 * HWP;
  float dy = ob[(size_t)(kk * 2 + 0) * HWP + p];
  float dx = ob[(size_t)(kk * 2 + 1) * HWP + p];
  float py = (float)(y + ki - 1) + dy;
  float px = (float)(x0c + kj - 1) + dx;
  float y0f = floorf(py), x0f = floorf(px);
  int y0 = (int)y0f, x0 = (int)x0f;
  float wy = py - y0f, wx = px - x0f;
  int   yi[2] = { y0, y0 + 1 }, xi[2] = { x0, x0 + 1 };
  float wyv[2] = { 1.f - wy, wy }, wxv[2] = { 1.f - wx, wx };
  float wgt[4]; int idx[4];
  #pragma unroll
  for (int cy = 0; cy < 2; ++cy)
    #pragma unroll
    for (int cx = 0; cx < 2; ++cx) {
      int yy = yi[cy], xq = xi[cx];
      bool valid = (yy >= 0) && (yy < HH) && (xq >= 0) && (xq < WWID);
      int ycl = yy < 0 ? 0 : (yy > HH - 1 ? HH - 1 : yy);
      int xcl = xq < 0 ? 0 : (xq > WWID - 1 ? WWID - 1 : xq);
      wgt[cy * 2 + cx] = valid ? wyv[cy] * wxv[cx] : 0.f;
      idx[cy * 2 + cx] = ycl * WWID + xcl;
    }
  const float* xb = x + (size_t)b * CIN * HWP;
  _Float16* dst = Ah + ((size_t)(b * HWP + p)) * KDIM + kk * CIN;
  #pragma unroll 4
  for (int c = 0; c < CIN; ++c) {
    const float* xp = xb + (size_t)c * HWP;
    float v = wgt[0] * xp[idx[0]] + wgt[1] * xp[idx[1]]
            + wgt[2] * xp[idx[2]] + wgt[3] * xp[idx[3]];
    dst[c] = (_Float16)v;
  }
}

// ------- weight permute+convert: w[o][c][kk] -> fp16 wh[o][kk*128+c] -------
__global__ void convw_k(const float* __restrict__ w, _Float16* __restrict__ wh) {
  int gid = blockIdx.x * blockDim.x + threadIdx.x;
  if (gid >= COUT * KDIM) return;
  int o = gid / KDIM; int r = gid % KDIM; int kk = r / CIN; int c = r % CIN;
  wh[gid] = (_Float16)w[(size_t)o * KDIM + c * 9 + kk];
}

// ------------- x NCHW -> fp16 [m=b*HW+p][c] (im2col for conv1x1) -------------
__global__ void x2ax_k(const float* __restrict__ x, _Float16* __restrict__ ax) {
  int m = blockIdx.x * blockDim.x + threadIdx.x;
  if (m >= MTOT) return;
  int b = m / HWP, p = m % HWP;
  const float* xb = x + (size_t)b * CIN * HWP + p;
  _Float16* d = ax + (size_t)m * CIN;
  #pragma unroll 4
  for (int c = 0; c < CIN; ++c) d[c] = (_Float16)xb[(size_t)c * HWP];
}

__global__ void f2h_k(const float* __restrict__ s, _Float16* __restrict__ d, int n) {
  int i = blockIdx.x * blockDim.x + threadIdx.x;
  if (i < n) d[i] = (_Float16)s[i];
}

__device__ __forceinline__ v16h make_afrag(const _Float16* row, int h) {
  v8h lo = *(const v8h*)(row + h * 8);
  v8h hi = *(const v8h*)(row + 16 + h * 8);
  return __builtin_shufflevector(lo, hi, 0,1,2,3,4,5,6,7,8,9,10,11,12,13,14,15);
}

// ---------------- WMMA GEMM: out[b][o][p] = A[m][K] * Wt[o][K] + bias ----------------
// 4 waves / block; block computes 128 M rows x 128 N.  B tile (32K x 128N fp16)
// is double-buffered through LDS and shared by all 4 waves; each wave owns
// 2 M-tiles (16 WMMAs per 32-wide k-step).
__global__ __launch_bounds__(128) void wmma_gemm_k(
    const _Float16* __restrict__ A, const _Float16* __restrict__ Wt,
    const float* __restrict__ bias, float* __restrict__ out, int Kd) {
  __shared__ _Float16 bs[2][128 * BSTR];   // 2 x 10 KB (padded rows)
  const int tid  = threadIdx.x;
  const int wave = tid >> 5;
  const int lane = tid & 31;
  const int r = lane & 15, h = lane >> 4;
  const int m0 = blockIdx.x * 128 + wave * 32;

  v8f acc[2][8] = {};
  const _Float16* arow0 = A + (size_t)(m0 + r) * Kd;
  const _Float16* arow1 = A + (size_t)(m0 + 16 + r) * Kd;

  // cooperative B staging: thread t copies the 32-half K-slice of column n=t
  auto stage = [&](int k, int buf) {
    const v8h* src = (const v8h*)(Wt + (size_t)tid * Kd + k);
    _Float16* d = &bs[buf][tid * BSTR];
    *(v8h*)(d)      = src[0];
    *(v8h*)(d + 8)  = src[1];
    *(v8h*)(d + 16) = src[2];
    *(v8h*)(d + 24) = src[3];
  };
  stage(0, 0);

  for (int k = 0; k < Kd; k += 32) {
    const int cur = (k >> 5) & 1;
    // this step's A fragments (global; independent of the barrier)
    v16h a0 = make_afrag(arow0 + k, h);
    v16h a1 = make_afrag(arow1 + k, h);
    __syncthreads();                       // bs[cur] fully staged
    if (k + 32 < Kd) {
      stage(k + 32, cur ^ 1);              // overlap next B tile with WMMA burst
      __builtin_prefetch(arow0 + k + 32, 0, 0);
      __builtin_prefetch(arow1 + k + 32, 0, 0);
    }
    #pragma unroll
    for (int nt = 0; nt < 8; ++nt) {
      const _Float16* bp = &bs[cur][(nt * 16 + r) * BSTR + h * 8];
      v8h blo = *(const v8h*)bp;
      v8h bhi = *(const v8h*)(bp + 16);
      v16h bf = __builtin_shufflevector(blo, bhi, 0,1,2,3,4,5,6,7,8,9,10,11,12,13,14,15);
      acc[0][nt] = __builtin_amdgcn_wmma_f32_16x16x32_f16(
          false, a0, false, bf, (short)0, acc[0][nt], false, false);
      acc[1][nt] = __builtin_amdgcn_wmma_f32_16x16x32_f16(
          false, a1, false, bf, (short)0, acc[1][nt], false, false);
    }
  }

  #pragma unroll
  for (int mt = 0; mt < 2; ++mt) {
    #pragma unroll
    for (int nt = 0; nt < 8; ++nt) {
      int o = nt * 16 + r;
      float bv = bias[o];
      #pragma unroll
      for (int v = 0; v < 8; ++v) {
        int m = m0 + mt * 16 + v + 8 * h;   // D: M = v + 8*(lane>>4), N = lane&15
        int b = m / HWP;
        int p = m - b * HWP;
        out[((size_t)(b * COUT + o)) * HWP + p] = acc[mt][nt][v] + bv;
      }
    }
  }
}

// ---------------- mean over HW: src[q*HWP + i] -> dst[(q/perB)*stride + q%perB + off] ----------------
__global__ void pool_mean_k(const float* __restrict__ src, float* __restrict__ dst,
                            int perB, int outStride, int chOff) {
  int q = blockIdx.x; int tid = threadIdx.x;
  __shared__ float red[256];
  const float* s = src + (size_t)q * HWP;
  float acc = 0.f;
  for (int i = tid; i < HWP; i += 256) acc += s[i];
  red[tid] = acc; __syncthreads();
  for (int o = 128; o > 0; o >>= 1) {
    if (tid < o) red[tid] += red[tid + o];
    __syncthreads();
  }
  if (tid == 0) dst[(q / perB) * outStride + (q % perB) + chOff] = red[0] * (1.f / HWP);
}

// ---------------- gating MLP: catpool[b][256] -> softmax weights wts[b][2] ----------------
__global__ void mlp_gate_k(const float* __restrict__ cp,
                           const float* __restrict__ w1, const float* __restrict__ b1,
                           const float* __restrict__ w2, const float* __restrict__ b2,
                           float* __restrict__ wts) {
  int b = blockIdx.x; int tid = threadIdx.x;  // 32 threads
  __shared__ float hbuf[8];
  __shared__ float zbuf[2];
  if (tid < 8) {
    float s = b1[tid];
    const float* wr = w1 + tid * 256;
    const float* cb = cp + b * 256;
    for (int c = 0; c < 256; ++c) s += wr[c] * cb[c];
    hbuf[tid] = s > 0.f ? s : 0.f;
  }
  __syncthreads();
  if (tid < 2) {
    float s = b2[tid];
    for (int j = 0; j < 8; ++j) s += w2[tid * 8 + j] * hbuf[j];
    zbuf[tid] = s;
  }
  __syncthreads();
  if (tid == 0) {
    float m = fmaxf(zbuf[0], zbuf[1]);
    float e0 = expf(zbuf[0] - m), e1 = expf(zbuf[1] - m);
    float inv = 1.f / (e0 + e1);
    wts[b * 2 + 0] = e0 * inv; wts[b * 2 + 1] = e1 * inv;
  }
}

// ---------------- global gate path: gx[b][128] -> gvec[b][128] ----------------
__global__ void mlp_g_k(const float* __restrict__ gx,
                        const float* __restrict__ w1, const float* __restrict__ b1,
                        const float* __restrict__ w2, const float* __restrict__ b2,
                        const float* __restrict__ w3, const float* __restrict__ b3,
                        float* __restrict__ gvec) {
  int b = blockIdx.x; int tid = threadIdx.x;  // 128 threads
  __shared__ float s_in[128], s_g1[64], s_g2[64];
  s_in[tid] = gx[b * 128 + tid];
  __syncthreads();
  if (tid < 64) {
    float s = b1[tid];
    for (int c = 0; c < 128; ++c) s += w1[tid * 128 + c] * s_in[c];
    s_g1[tid] = s > 0.f ? s : 0.f;
  }
  __syncthreads();
  if (tid < 64) {
    float s = b2[tid];
    for (int c = 0; c < 64; ++c) s += w2[tid * 64 + c] * s_g1[c];
    s_g2[tid] = s > 0.f ? s : 0.f;
  }
  __syncthreads();
  {
    float s = b3[tid];
    for (int c = 0; c < 64; ++c) s += w3[tid * 64 + c] * s_g2[c];
    gvec[b * 128 + tid] = 1.f / (1.f + expf(-s));
  }
}

// ---------------- final fusion ----------------
__global__ void fuse_k(const float* __restrict__ s0, const float* __restrict__ s1,
                       const float* __restrict__ xr, const float* __restrict__ wts,
                       const float* __restrict__ gvec, float* __restrict__ out) {
  int i = blockIdx.x * blockDim.x + threadIdx.x;
  if (i >= BATCH * COUT * HWP) return;
  int b = i / (COUT * HWP);
  int o = (i / HWP) % COUT;
  out[i] = wts[b * 2 + 0] * s0[i] + wts[b * 2 + 1] * s1[i] + xr[i] * gvec[b * 128 + o];
}

extern "C" void kernel_launch(void* const* d_in, const int* in_sizes, int n_in,
                              void* d_out, int out_size, void* d_ws, size_t ws_size,
                              hipStream_t stream) {
  (void)in_sizes; (void)n_in; (void)out_size; (void)ws_size;
  const float* x      = (const float*)d_in[0];
  const float* off_w0 = (const float*)d_in[1];
  const float* off_b0 = (const float*)d_in[2];
  const float* dw0    = (const float*)d_in[3];
  const float* db0    = (const float*)d_in[4];
  const float* off_w1 = (const float*)d_in[5];
  const float* off_b1 = (const float*)d_in[6];
  const float* dw1    = (const float*)d_in[7];
  const float* db1    = (const float*)d_in[8];
  const float* wg_w1  = (const float*)d_in[9];
  const float* wg_b1  = (const float*)d_in[10];
  const float* wg_w2  = (const float*)d_in[11];
  const float* wg_b2  = (const float*)d_in[12];
  const float* gp_w1  = (const float*)d_in[13];
  const float* gp_b1  = (const float*)d_in[14];
  const float* gp_w2  = (const float*)d_in[15];
  const float* gp_b2  = (const float*)d_in[16];
  const float* gp_w3  = (const float*)d_in[17];
  const float* gp_b3  = (const float*)d_in[18];
  const float* cr_w   = (const float*)d_in[19];
  const float* cr_b   = (const float*)d_in[20];
  float* out = (float*)d_out;

  // carve workspace (256B aligned slices)
  char* wsp = (char*)d_ws;
  auto carve = [&](size_t bytes) -> void* {
    void* p = (void*)wsp;
    wsp += (bytes + 255) & ~(size_t)255;
    return p;
  };
  float*    s0    = (float*)   carve((size_t)BATCH * COUT * HWP * 4);   // 18.9MB
  float*    s1    = (float*)   carve((size_t)BATCH * COUT * HWP * 4);   // 18.9MB
  float*    xr    = (float*)   carve((size_t)BATCH * COUT * HWP * 4);   // 18.9MB
  float*    offs  = (float*)   carve((size_t)BATCH * 18 * HWP * 4);     // 2.65MB
  _Float16* Ah    = (_Float16*)carve((size_t)MTOT * KDIM * 2);          // 84.9MB
  _Float16* Ax    = (_Float16*)carve((size_t)MTOT * CIN * 2);           // 9.4MB
  _Float16* Wh    = (_Float16*)carve((size_t)COUT * KDIM * 2);          // 288KB
  _Float16* Whcr  = (_Float16*)carve((size_t)COUT * CIN * 2);           // 32KB
  float*    catpool = (float*) carve((size_t)BATCH * 256 * 4);
  float*    gx    = (float*)   carve((size_t)BATCH * 128 * 4);
  float*    wts   = (float*)   carve((size_t)BATCH * 2 * 4);
  float*    gvec  = (float*)   carve((size_t)BATCH * 128 * 4);

  // ---- conv1x1 channel-reduce path (WMMA GEMM, K=128) ----
  f2h_k<<<(COUT * CIN + 255) / 256, 256, 0, stream>>>(cr_w, Whcr, COUT * CIN);
  x2ax_k<<<(MTOT + 127) / 128, 128, 0, stream>>>(x, Ax);
  wmma_gemm_k<<<MTOT / 128, 128, 0, stream>>>(Ax, Whcr, cr_b, xr, CIN);

  // ---- global gating path ----
  pool_mean_k<<<BATCH * CIN, 256, 0, stream>>>(x, gx, 128, 128, 0);
  mlp_g_k<<<BATCH, 128, 0, stream>>>(gx, gp_w1, gp_b1, gp_w2, gp_b2, gp_w3, gp_b3, gvec);

  // ---- deformable branch 0 (offsets -> sample/im2col -> WMMA GEMM, K=1152) ----
  conv_off_k<<<(BATCH * 18 * HWP + 255) / 256, 256, 0, stream>>>(x, off_w0, off_b0, offs);
  sample_k<<<(BATCH * KK2 * HWP + 255) / 256, 256, 0, stream>>>(x, offs, Ah);
  convw_k<<<(COUT * KDIM + 255) / 256, 256, 0, stream>>>(dw0, Wh);
  wmma_gemm_k<<<MTOT / 128, 128, 0, stream>>>(Ah, Wh, db0, s0, KDIM);

  // ---- deformable branch 1 (reuses offs/Ah/Wh scratch; stream-ordered) ----
  conv_off_k<<<(BATCH * 18 * HWP + 255) / 256, 256, 0, stream>>>(x, off_w1, off_b1, offs);
  sample_k<<<(BATCH * KK2 * HWP + 255) / 256, 256, 0, stream>>>(x, offs, Ah);
  convw_k<<<(COUT * KDIM + 255) / 256, 256, 0, stream>>>(dw1, Wh);
  wmma_gemm_k<<<MTOT / 128, 128, 0, stream>>>(Ah, Wh, db1, s1, KDIM);

  // ---- pooled gating + fusion ----
  pool_mean_k<<<BATCH * COUT, 256, 0, stream>>>(s0, catpool, 128, 256, 0);
  pool_mean_k<<<BATCH * COUT, 256, 0, stream>>>(s1, catpool, 128, 256, 128);
  mlp_gate_k<<<BATCH, 32, 0, stream>>>(catpool, wg_w1, wg_b1, wg_w2, wg_b2, wts);
  fuse_k<<<(BATCH * COUT * HWP + 255) / 256, 256, 0, stream>>>(s0, s1, xr, wts, gvec, out);
}
